// Decoder_3083786518741
// MI455X (gfx1250) — compile-verified, matrix operand order checked by reading, not run on hardware
//
#include <hip/hip_runtime.h>
#include <hip/hip_bf16.h>
#include <math.h>

// ---------------- problem dims ----------------
#define B_  256
#define S_  128
#define T_  4
#define H_  512
#define E_  256
#define V_  50000
#define X_  50
#define H2_ 1024   // 2H
#define H4_ 2048   // 4H
#define VX_ 50050  // V+X

typedef __attribute__((ext_vector_type(16))) __bf16 v16bf;
typedef __attribute__((ext_vector_type(8)))  float  v8f;

// ---------------- segmented row-major matrix (concat along K) ----------------
// All segment widths used here are multiples of 64, and K-runs are 64-aligned,
// so the double-buffered 2x32 K-step never straddles a segment boundary.
struct Segs3 {
    const float* p0; int w0;
    const float* p1; int w1;
    const float* p2; int w2;
};

// Resolve the segment containing k0: returns base ptr, row width, local k, remaining.
__device__ __forceinline__ void seg_find(const Segs3& s, int k0, const float*& base,
                                         int& w, int& loc, int& rem) {
    if (k0 < s.w0) { base = s.p0; w = s.w0; loc = k0; rem = s.w0 - k0; return; }
    k0 -= s.w0;
    if (k0 < s.w1) { base = s.p1; w = s.w1; loc = k0; rem = s.w1 - k0; return; }
    k0 -= s.w1;
    base = s.p2; w = s.w2; loc = k0; rem = s.w2 - k0;
}

__device__ __forceinline__ v16bf make_frag(float4 x0, float4 x1, float4 y0, float4 y1) {
    v16bf f;
    f[0]  = (__bf16)x0.x; f[1]  = (__bf16)x0.y; f[2]  = (__bf16)x0.z; f[3]  = (__bf16)x0.w;
    f[4]  = (__bf16)x1.x; f[5]  = (__bf16)x1.y; f[6]  = (__bf16)x1.z; f[7]  = (__bf16)x1.w;
    f[8]  = (__bf16)y0.x; f[9]  = (__bf16)y0.y; f[10] = (__bf16)y0.z; f[11] = (__bf16)y0.w;
    f[12] = (__bf16)y1.x; f[13] = (__bf16)y1.y; f[14] = (__bf16)y1.z; f[15] = (__bf16)y1.w;
    return f;
}

// Raw fp32 data for one 32-wide K chunk: W frag + 4 A frags (20 x float4).
struct RawBuf {
    float4 w[4];
    float4 a[4][4];
};

__device__ __forceinline__ void load_chunk(RawBuf& b, const float* wp,
                                           const float* const ap[4], int kk) {
    b.w[0] = *(const float4*)(wp + kk);
    b.w[1] = *(const float4*)(wp + kk + 4);
    b.w[2] = *(const float4*)(wp + kk + 16);
    b.w[3] = *(const float4*)(wp + kk + 20);
#pragma unroll
    for (int mt = 0; mt < 4; ++mt) {
        b.a[mt][0] = *(const float4*)(ap[mt] + kk);
        b.a[mt][1] = *(const float4*)(ap[mt] + kk + 4);
        b.a[mt][2] = *(const float4*)(ap[mt] + kk + 16);
        b.a[mt][3] = *(const float4*)(ap[mt] + kk + 20);
    }
}

__device__ __forceinline__ void compute_chunk(const RawBuf& b, v8f acc[4]) {
    v16bf wf = make_frag(b.w[0], b.w[1], b.w[2], b.w[3]);
#pragma unroll
    for (int mt = 0; mt < 4; ++mt) {
        v16bf af = make_frag(b.a[mt][0], b.a[mt][1], b.a[mt][2], b.a[mt][3]);
        acc[mt] = __builtin_amdgcn_wmma_f32_16x16x32_bf16(false, af, false, wf,
                                                          (short)0, acc[mt], false, false);
    }
}

// ---------------- generic WMMA GEMM: out = A @ W^T + b1 + b2 ----------------
// grid = (N/16, M/64), block = 32; each wave owns 4 M-tiles, shares the W frag.
// Register double-buffering: chunk i+1's 20 loads are in flight while chunk i
// is converted and fed to the 4 WMMAs (loop carries the data dependence).
__global__ __launch_bounds__(32) __attribute__((amdgpu_waves_per_eu(4)))
void gemm_wmma(
    Segs3 A, Segs3 W, const float* __restrict__ b1, const float* __restrict__ b2,
    float* __restrict__ out, int K, int ldOut)
{
    const int lane  = threadIdx.x & 31;
    const int r15   = lane & 15;
    const int g     = (lane & 16) ? 8 : 0;
    const int nBase = blockIdx.x * 16;
    const int mBase = blockIdx.y * 64;
    v8f acc[4] = {{}, {}, {}, {}};

    int k0 = 0;
    while (k0 < K) {
        const float *aBase, *wBase;
        int aW, aLoc, remA, wW, wLoc, remW;
        seg_find(A, k0, aBase, aW, aLoc, remA);
        seg_find(W, k0, wBase, wW, wLoc, remW);
        const float* ap[4];
#pragma unroll
        for (int mt = 0; mt < 4; ++mt)
            ap[mt] = aBase + (size_t)(mBase + mt * 16 + r15) * aW + aLoc + g;
        const float* wp = wBase + (size_t)(nBase + r15) * wW + wLoc + g;
        const int len = (remA < remW ? remA : remW);   // always a multiple of 64 here

        RawBuf buf0, buf1;
        load_chunk(buf0, wp, ap, 0);
        for (int kk = 0; kk < len; kk += 64) {
            load_chunk(buf1, wp, ap, kk + 32);
            compute_chunk(buf0, acc);
            if (kk + 64 < len) load_chunk(buf0, wp, ap, kk + 64);
            compute_chunk(buf1, acc);
        }
        k0 += len;
    }

    const int col = nBase + r15;
    const float bias = (b1 ? b1[col] : 0.f) + (b2 ? b2[col] : 0.f);
#pragma unroll
    for (int mt = 0; mt < 4; ++mt) {
        const int rbase = mBase + mt * 16 + ((lane & 16) ? 8 : 0);
#pragma unroll
        for (int r = 0; r < 8; ++r)
            out[(size_t)(rbase + r) * ldOut + col] = acc[mt][r] + bias;
    }
}

// ---------------- fused encoder-attention scorer ----------------
// et[b,s] += sum_n v_e[n] * tanh( (enc_out[b,s,:] @ Wh_e^T)[n] + qe[b,n] )
// grid = (1024/64, 32768/64), block = 256 (8 waves), 64x64 C tile / block.
__global__ __launch_bounds__(256) void enc_att_kernel(
    const float* __restrict__ enc_out,  // (B*S, 1024)
    const float* __restrict__ Wh_e,     // (1024, 1024)
    const float* __restrict__ qe,       // (B, 1024)
    const float* __restrict__ v_e,      // (1024)
    float* __restrict__ et)             // (B*S) atomic accumulate
{
    __shared__ __align__(32) __bf16 lA[4][32][16];
    __shared__ __align__(32) __bf16 lW[4][32][16];

    const int lane = threadIdx.x & 31;
    const int wave = threadIdx.x >> 5;   // 0..7
    const int wm   = wave >> 2;          // 0..1  (32 rows each)
    const int wn   = wave & 3;           // 0..3  (16 cols each)
    const int mBase = blockIdx.y * 64;
    const int nBase = blockIdx.x * 64;

    v8f acc0 = {}, acc1 = {};
    for (int k0 = 0; k0 < H2_; k0 += 32) {
        __syncthreads();
        {   // cooperative swizzled fill: 128 A slots + 128 W slots
            const int t    = threadIdx.x;
            const bool isW = t >= 128;
            const int slot = t & 127;
            const int tile = slot >> 5;     // 0..3
            const int ln   = slot & 31;
            const int row  = (isW ? nBase : mBase) + tile * 16 + (ln & 15);
            const float* src = (isW ? Wh_e : enc_out) + (size_t)row * H2_ + k0;
            __builtin_prefetch(src + 64, 0, 1);
            const int g = (ln & 16) ? 8 : 0;
            float4 x0 = *(const float4*)(src + g);
            float4 x1 = *(const float4*)(src + g + 4);
            float4 y0 = *(const float4*)(src + g + 16);
            float4 y1 = *(const float4*)(src + g + 20);
            __bf16* dst = isW ? &lW[tile][ln][0] : &lA[tile][ln][0];
            *(v16bf*)dst = make_frag(x0, x1, y0, y1);
        }
        __syncthreads();
        v16bf a0 = *(const v16bf*)&lA[wm * 2 + 0][lane][0];
        v16bf a1 = *(const v16bf*)&lA[wm * 2 + 1][lane][0];
        v16bf w  = *(const v16bf*)&lW[wn][lane][0];
        acc0 = __builtin_amdgcn_wmma_f32_16x16x32_bf16(false, a0, false, w, (short)0, acc0, false, false);
        acc1 = __builtin_amdgcn_wmma_f32_16x16x32_bf16(false, a1, false, w, (short)0, acc1, false, false);
    }

    // epilogue: tanh(+qe)*v_e, reduce over the 16 columns this wave owns
    const int b    = mBase >> 7;                        // 64-row blocks never span b (S=128)
    const int col  = nBase + wn * 16 + (lane & 15);
    const float ve = v_e[col];
    const float q  = qe[(size_t)b * H2_ + col];
    const int half = (lane & 16) ? 8 : 0;
#pragma unroll
    for (int pair = 0; pair < 2; ++pair) {
        v8f acc = pair ? acc1 : acc0;
#pragma unroll
        for (int r = 0; r < 8; ++r) {
            float v = tanhf(acc[r] + q) * ve;
            v += __shfl_xor(v, 1, 32);
            v += __shfl_xor(v, 2, 32);
            v += __shfl_xor(v, 4, 32);
            v += __shfl_xor(v, 8, 32);
            if ((lane & 15) == 0) {
                int row = mBase + wm * 32 + pair * 16 + r + half;
                atomicAdd(&et[row], v);
            }
        }
    }
}

// ---------------- small kernels ----------------
__global__ void zero_kernel(float* p, int n) {
    int i = blockIdx.x * blockDim.x + threadIdx.x;
    if (i < n) p[i] = 0.f;
}

__device__ __forceinline__ float sigmoidf_(float x) { return 1.f / (1.f + expf(-x)); }

__global__ void lstm_point(const float* __restrict__ gates, const float* __restrict__ c0,
                           float* __restrict__ h, float* __restrict__ c,
                           float* __restrict__ oh, float* __restrict__ oc) {
    int i = blockIdx.x * blockDim.x + threadIdx.x;   // B*H
    if (i >= B_ * H_) return;
    int b = i >> 9, j = i & (H_ - 1);
    const float* g = gates + (size_t)b * H4_;
    float ig = g[j], fg = g[H_ + j], gg = g[2 * H_ + j], og = g[3 * H_ + j];
    float cc = sigmoidf_(fg) * c0[i] + sigmoidf_(ig) * tanhf(gg);
    float hh = sigmoidf_(og) * tanhf(cc);
    h[i] = hh; c[i] = cc; oh[i] = hh; oc[i] = cc;
}

__global__ void enc_soft(const float* __restrict__ et, const float* __restrict__ sumts,
                         const float* __restrict__ mask, float* __restrict__ at,
                         float* __restrict__ sum_new) {
    int b = blockIdx.x, s = threadIdx.x;             // 128 threads
    int i = b * S_ + s;
    float ex = expf(et[i]);
    float st = sumts[i];
    sum_new[i] = st + ex;
    float a = (ex / st) * mask[i];
    __shared__ float red[S_];
    red[s] = a; __syncthreads();
    for (int off = 64; off > 0; off >>= 1) { if (s < off) red[s] += red[s + off]; __syncthreads(); }
    at[i] = a / red[0];
}

__global__ void ct_e_kernel(const float* __restrict__ at, const float* __restrict__ enc_out,
                            float* __restrict__ cte, float* __restrict__ cte_out) {
    int b = blockIdx.x;
    __shared__ float A[S_];
    if (threadIdx.x < S_) {
        // CDNA5 async global->LDS copy (ASYNCcnt tracked)
        unsigned ldsa = (unsigned)(uintptr_t)&A[threadIdx.x];
        const float* ga = at + (size_t)b * S_ + threadIdx.x;
        asm volatile("global_load_async_to_lds_b32 %0, %1, off"
                     :: "v"(ldsa), "v"(ga) : "memory");
    }
    asm volatile("s_wait_asynccnt 0" ::: "memory");
    __syncthreads();
    for (int n = threadIdx.x; n < H2_; n += 256) {
        const float* e = enc_out + (size_t)b * S_ * H2_ + n;
        float acc = 0.f;
        for (int s = 0; s < S_; ++s) acc += A[s] * e[(size_t)s * H2_];
        cte[(size_t)b * H2_ + n] = acc;
        cte_out[(size_t)b * H2_ + n] = acc;
    }
}

__global__ void dec_att(const float* __restrict__ P, const float* __restrict__ qd,
                        const float* __restrict__ v_d, const float* __restrict__ prev_s,
                        float* __restrict__ ct_d) {
    int b = blockIdx.x, t = threadIdx.x;             // 128 threads
    float p[T_] = {0.f, 0.f, 0.f, 0.f};
    for (int n = t; n < H_; n += 128) {
        float q = qd[(size_t)b * H_ + n], v = v_d[n];
#pragma unroll
        for (int tt = 0; tt < T_; ++tt)
            p[tt] += tanhf(P[((size_t)b * T_ + tt) * H_ + n] + q) * v;
    }
    __shared__ float red[T_][128];
#pragma unroll
    for (int tt = 0; tt < T_; ++tt) red[tt][t] = p[tt];
    __syncthreads();
    for (int off = 64; off > 0; off >>= 1) {
        if (t < off) {
#pragma unroll
            for (int tt = 0; tt < T_; ++tt) red[tt][t] += red[tt][t + off];
        }
        __syncthreads();
    }
    __shared__ float ad[T_];
    if (t == 0) {
        float m = fmaxf(fmaxf(red[0][0], red[1][0]), fmaxf(red[2][0], red[3][0]));
        float s = 0.f, e[T_];
#pragma unroll
        for (int tt = 0; tt < T_; ++tt) { e[tt] = expf(red[tt][0] - m); s += e[tt]; }
#pragma unroll
        for (int tt = 0; tt < T_; ++tt) ad[tt] = e[tt] / s;
    }
    __syncthreads();
    for (int n = t; n < H_; n += 128) {
        float acc = 0.f;
#pragma unroll
        for (int tt = 0; tt < T_; ++tt)
            acc += ad[tt] * prev_s[((size_t)b * T_ + tt) * H_ + n];
        ct_d[(size_t)b * H_ + n] = acc;
    }
}

__global__ void pgen_kernel(const float* __restrict__ cte, const float* __restrict__ ctd,
                            const float* __restrict__ h, const float* __restrict__ c,
                            const float* __restrict__ x, const float* __restrict__ Wpg,
                            const float* __restrict__ bpg, float* __restrict__ pg) {
    int b = blockIdx.x, t = threadIdx.x;             // 256 threads, K = 2816
    float acc = 0.f;
    for (int k = t; k < 5 * H_ + E_; k += 256) {
        float val;
        if (k < H2_)                val = cte[(size_t)b * H2_ + k];
        else if (k < H2_ + H_)      val = ctd[(size_t)b * H_ + (k - H2_)];
        else if (k < H2_ + 2 * H_)  val = h[(size_t)b * H_ + (k - H2_ - H_)];
        else if (k < H2_ + 3 * H_)  val = c[(size_t)b * H_ + (k - H2_ - 2 * H_)];
        else                        val = x[(size_t)b * E_ + (k - H2_ - 3 * H_)];
        acc += val * Wpg[k];
    }
    __shared__ float red[256];
    red[t] = acc; __syncthreads();
    for (int off = 128; off > 0; off >>= 1) { if (t < off) red[t] += red[t + off]; __syncthreads(); }
    if (t == 0) pg[b] = sigmoidf_(red[0] + bpg[0]);
}

__global__ void rowstats(const float* __restrict__ logits, float* __restrict__ rmax,
                         float* __restrict__ rsum) {
    int b = blockIdx.x, t = threadIdx.x;             // 256 threads
    const float* row = logits + (size_t)b * VX_;
    __shared__ float red[256];
    float m = -INFINITY;
    for (int v = t; v < V_; v += 256) m = fmaxf(m, row[v]);
    red[t] = m; __syncthreads();
    for (int off = 128; off > 0; off >>= 1) { if (t < off) red[t] = fmaxf(red[t], red[t + off]); __syncthreads(); }
    float mx = red[0]; __syncthreads();
    float s = 0.f;
    for (int v = t; v < V_; v += 256) s += expf(row[v] - mx);
    red[t] = s; __syncthreads();
    for (int off = 128; off > 0; off >>= 1) { if (t < off) red[t] += red[t + off]; __syncthreads(); }
    if (t == 0) { rmax[b] = mx; rsum[b] = red[0]; }
}

__global__ void finalize(float* __restrict__ out, const float* __restrict__ rmax,
                         const float* __restrict__ rsum, const float* __restrict__ pg) {
    size_t i = (size_t)blockIdx.x * 256 + threadIdx.x;
    if (i >= (size_t)B_ * VX_) return;
    int b = (int)(i / VX_), v = (int)(i % VX_);
    float val = 0.f;
    if (v < V_) val = pg[b] * expf(out[i] - rmax[b]) / rsum[b];
    out[i] = val;
}

__global__ void scatter_kernel(float* __restrict__ out, const int* __restrict__ idx,
                               const float* __restrict__ at, const float* __restrict__ pg) {
    int i = blockIdx.x * 256 + threadIdx.x;          // B*S
    if (i >= B_ * S_) return;
    int b = i >> 7;
    atomicAdd(&out[(size_t)b * VX_ + idx[i]], (1.f - pg[b]) * at[i]);
}

__global__ void prevs_kernel(const float* __restrict__ prev_s, const float* __restrict__ h,
                             float* __restrict__ out) {
    int i = blockIdx.x * 256 + threadIdx.x;          // B*(T+1)*H
    if (i >= B_ * (T_ + 1) * H_) return;
    int b = i / ((T_ + 1) * H_);
    int rem = i - b * (T_ + 1) * H_;
    int t = rem / H_, j = rem - t * H_;
    out[i] = (t < T_) ? prev_s[((size_t)b * T_ + t) * H_ + j] : h[(size_t)b * H_ + j];
}

// ---------------- launcher ----------------
static inline Segs3 seg1(const float* p, int w) { return Segs3{p, w, nullptr, 0, nullptr, 0}; }
static inline Segs3 seg2(const float* p0, int w0, const float* p1, int w1) { return Segs3{p0, w0, p1, w1, nullptr, 0}; }
static inline Segs3 seg3(const float* p0, int w0, const float* p1, int w1, const float* p2, int w2) { return Segs3{p0, w0, p1, w1, p2, w2}; }

extern "C" void kernel_launch(void* const* d_in, const int* in_sizes, int n_in,
                              void* d_out, int out_size, void* d_ws, size_t ws_size,
                              hipStream_t stream) {
    const float* x_t    = (const float*)d_in[0];
    const float* h0     = (const float*)d_in[1];
    const float* c0     = (const float*)d_in[2];
    const float* enc    = (const float*)d_in[3];
    const float* mask   = (const float*)d_in[4];
    const float* ct_e0  = (const float*)d_in[5];
    const float* sumts  = (const float*)d_in[7];
    const float* prev_s = (const float*)d_in[8];
    const float* Wxc    = (const float*)d_in[9];
    const float* bxc    = (const float*)d_in[10];
    const float* W_ih   = (const float*)d_in[11];
    const float* W_hh   = (const float*)d_in[12];
    const float* b_ih   = (const float*)d_in[13];
    const float* b_hh   = (const float*)d_in[14];
    const float* Wh_e   = (const float*)d_in[15];
    const float* Ws_e   = (const float*)d_in[16];
    const float* bs_e   = (const float*)d_in[17];
    const float* v_e    = (const float*)d_in[18];
    const float* Wprev_d= (const float*)d_in[19];
    const float* Ws_d   = (const float*)d_in[20];
    const float* bs_d   = (const float*)d_in[21];
    const float* v_d    = (const float*)d_in[22];
    const float* Wpg    = (const float*)d_in[23];
    const float* bpg    = (const float*)d_in[24];
    const float* WV     = (const float*)d_in[25];
    const float* bV     = (const float*)d_in[26];
    const float* WV1    = (const float*)d_in[27];
    const float* bV1    = (const float*)d_in[28];
    const int*   ebev   = (const int*)d_in[29];

    float* out = (float*)d_out;
    float* ws  = (float*)d_ws;

    // workspace layout (floats)
    float* w_x     = ws;                     // B*E
    float* w_gates = w_x     + B_ * E_;      // B*4H
    float* w_h     = w_gates + B_ * H4_;     // B*H
    float* w_c     = w_h     + B_ * H_;      // B*H
    float* w_qe    = w_c     + B_ * H_;      // B*2H
    float* w_et    = w_qe    + B_ * H2_;     // B*S
    float* w_at    = w_et    + B_ * S_;      // B*S
    float* w_cte   = w_at    + B_ * S_;      // B*2H
    float* w_qd    = w_cte   + B_ * H2_;     // B*H
    float* w_P     = w_qd    + B_ * H_;      // B*T*H
    float* w_ctd   = w_P     + B_ * T_ * H_; // B*H
    float* w_pg    = w_ctd   + B_ * H_;      // B
    float* w_ov    = w_pg    + B_;           // B*H
    float* w_rmax  = w_ov    + B_ * H_;      // B
    float* w_rsum  = w_rmax  + B_;           // B

    // output layout (floats)
    float* o_final = out;                                // B*(V+X)
    float* o_h     = o_final + (size_t)B_ * VX_;         // B*H
    float* o_c     = o_h     + B_ * H_;                  // B*H
    float* o_cte   = o_c     + B_ * H_;                  // B*2H
    float* o_sum   = o_cte   + B_ * H2_;                 // B*S
    float* o_prevs = o_sum   + B_ * S_;                  // B*(T+1)*H

    // 1) zero the atomic accumulator for et
    zero_kernel<<<(B_ * S_ + 255) / 256, 256, 0, stream>>>(w_et, B_ * S_);

    // 2) x = [x_t | ct_e] @ Wxc^T + bxc           (256x256, K=1280)
    gemm_wmma<<<dim3(E_ / 16, B_ / 64), 32, 0, stream>>>(
        seg2(x_t, E_, ct_e0, H2_), seg1(Wxc, H2_ + E_), bxc, nullptr, w_x, H2_ + E_, E_);

    // 3) gates = [x | h0] @ [W_ih | W_hh]^T + b_ih + b_hh   (256x2048, K=768)
    gemm_wmma<<<dim3(H4_ / 16, B_ / 64), 32, 0, stream>>>(
        seg2(w_x, E_, h0, H_), seg2(W_ih, E_, W_hh, H_), b_ih, b_hh, w_gates, E_ + H_, H4_);

    // 4) LSTM nonlinearity -> h, c (ws + outputs)
    lstm_point<<<(B_ * H_ + 255) / 256, 256, 0, stream>>>(w_gates, c0, w_h, w_c, o_h, o_c);

    // 5) qe = [h | c] @ Ws_e^T + bs_e             (256x1024, K=1024)
    gemm_wmma<<<dim3(H2_ / 16, B_ / 64), 32, 0, stream>>>(
        seg2(w_h, H_, w_c, H_), seg1(Ws_e, H2_), bs_e, nullptr, w_qe, H2_, H2_);

    // 6) fused encoder attention scores -> et (atomic)
    enc_att_kernel<<<dim3(H2_ / 64, (B_ * S_) / 64), 256, 0, stream>>>(enc, Wh_e, w_qe, v_e, w_et);

    // 7) temporal softmax: at, sum_new
    enc_soft<<<B_, S_, 0, stream>>>(w_et, sumts, mask, w_at, o_sum);

    // 8) ct_e_new = at @ enc_out
    ct_e_kernel<<<B_, 256, 0, stream>>>(w_at, enc, w_cte, o_cte);

    // 9) qd = h @ Ws_d^T + bs_d                   (256x512, K=512)
    gemm_wmma<<<dim3(H_ / 16, B_ / 64), 32, 0, stream>>>(
        seg1(w_h, H_), seg1(Ws_d, H_), bs_d, nullptr, w_qd, H_, H_);

    // 10) P = prev_s @ Wprev_d^T                  (1024x512, K=512)
    gemm_wmma<<<dim3(H_ / 16, (B_ * T_) / 64), 32, 0, stream>>>(
        seg1(prev_s, H_), seg1(Wprev_d, H_), nullptr, nullptr, w_P, H_, H_);

    // 11) intra-decoder attention -> ct_d
    dec_att<<<B_, 128, 0, stream>>>(w_P, w_qd, v_d, prev_s, w_ctd);

    // 12) p_gen
    pgen_kernel<<<B_, 256, 0, stream>>>(w_cte, w_ctd, w_h, w_c, w_x, Wpg, bpg, w_pg);

    // 13) out_vec = [h | ct_e | ct_d] @ WV^T + bV (256x512, K=2048)
    gemm_wmma<<<dim3(H_ / 16, B_ / 64), 32, 0, stream>>>(
        seg3(w_h, H_, w_cte, H2_, w_ctd, H_), seg1(WV, H4_), bV, nullptr, w_ov, H4_, H_);

    // 14) logits = out_vec @ WV1^T + bV1 -> d_out rows (ld = 50050)
    gemm_wmma<<<dim3(V_ / 16, B_ / 64), 32, 0, stream>>>(
        seg1(w_ov, H_), seg1(WV1, H_), bV1, nullptr, o_final, H_, VX_);

    // 15-17) softmax over V, p_gen scaling, copy-attention scatter
    rowstats<<<B_, 256, 0, stream>>>(o_final, w_rmax, w_rsum);
    finalize<<<(int)(((size_t)B_ * VX_ + 255) / 256), 256, 0, stream>>>(o_final, w_rmax, w_rsum, w_pg);
    scatter_kernel<<<(B_ * S_ + 255) / 256, 256, 0, stream>>>(o_final, ebev, w_at, w_pg);

    // 18) prev_s_new = concat(prev_s, h)
    prevs_kernel<<<(B_ * (T_ + 1) * H_ + 255) / 256, 256, 0, stream>>>(prev_s, w_h, o_prevs);
}
